// NaiveFourierKANLayer_54262616818097
// MI455X (gfx1250) — compile-verified, matrix operand order checked by reading, not run on hardware
//
#include <hip/hip_runtime.h>
#include <hip/hip_bf16.h>

typedef __attribute__((ext_vector_type(16))) _Float16 v16h;
typedef __attribute__((ext_vector_type(8)))  _Float16 v8h;
typedef __attribute__((ext_vector_type(4)))  _Float16 v4h;
typedef __attribute__((ext_vector_type(8)))  float    v8f;
typedef __attribute__((ext_vector_type(4)))  float    v4f;

#define IDIM    256
#define ODIM    256
#define GRIDSZ  300
#define BATCH   1024
#define IG      (IDIM * GRIDSZ)      // 76800
#define KTOT    (2 * IG)             // 153600
#define KSPLIT  32
#define KSEG    (KTOT / KSPLIT)      // 4800 (multiple of 32; t uniform per block)
#define KCH     32                   // WMMA K per step
#define NCH     (KSEG / KCH)         // 150 chunks
#define MTILE   128
#define MSPLIT  (BATCH / MTILE)      // 8
#define NTHREADS 256                 // 8 waves (wave32)

// out[b,j] = bias[j]  (accumulation target for the K-split partials)
__global__ void fkan_bias_init(const float* __restrict__ bias, float* __restrict__ out) {
    const int idx = blockIdx.x * blockDim.x + threadIdx.x;
    out[idx] = bias[idx & (ODIM - 1)];
}

__global__ __launch_bounds__(NTHREADS)
void fkan_wmma(const float* __restrict__ x, const float* __restrict__ fc,
               float* __restrict__ out) {
    // Double-buffered B tile, stored pre-swizzled into WMMA B-fragment order:
    // for column j, halves [j*32 .. j*32+15] are the lane<16 fragment (K {0..7,16..23}),
    // halves [j*32+16 .. j*32+31] are the lane>=16 fragment (K {8..15,24..31}).
    __shared__ _Float16 bs[2][KCH * ODIM];   // 2 x 16 KB

    const int tid  = threadIdx.x;
    const int lane = tid & 31;
    const int wave = tid >> 5;                       // 0..7
    const int mblk = blockIdx.x & (MSPLIT - 1);      // 0..7
    const int kblk = blockIdx.x >> 3;                // 0..31
    const int ks   = kblk * KSEG;
    const int t    = ks / IG;                        // 0 = cos half, 1 = sin half (uniform)
    const int rseg = ks - t * IG;                    // offset within (i,g) space
    const float shift = t ? 1.5707963267948966f : 0.0f;   // sin(a) = cos(a - pi/2)
    const float* __restrict__ fcT = fc + (long)t * ODIM * IG;

    // staging decomposition: tid -> (kk4 = 4 consecutive K, jg = 8 j-rows)
    const int kk4 = (tid & 7) << 2;                  // 0,4,...,28
    const int jg  = tid >> 3;                        // 0..31
    // swizzle a K-position into fragment order (4-aligned chunks stay intact)
    const int s4  = (kk4 & 7) | ((kk4 & 16) >> 1) | ((kk4 & 8) << 1);

    const int laneHalf = (lane >> 4) << 3;           // 0 or 8 (K-offset of this lane group)
    const int rowA = mblk * MTILE + wave * 16 + (lane & 15);
    const float* __restrict__ xrow = x + rowA * IDIM;

    v8f acc[16];
    #pragma unroll
    for (int n = 0; n < 16; ++n) acc[n] = (v8f)0.0f;

    // ---- staging split in two phases so global-load latency is hidden by the
    // ---- WMMA sweep and the WMMA-side s_wait_dscnt only sees fragment loads.
    v4f sreg[8];                                     // raw f32 weight slice in flight
    auto stage_load = [&](int it) {                  // phase 1: global -> VGPR (b128)
        const int r0 = rseg + it * KCH;
        const float* src = fcT + r0 + kk4;
        #pragma unroll
        for (int jj = 0; jj < 8; ++jj)
            sreg[jj] = *(const v4f*)(src + (long)(jg * 8 + jj) * IG);
    };
    auto stage_store = [&](int buf) {                // phase 2: cvt + VGPR -> LDS (b64)
        #pragma unroll
        for (int jj = 0; jj < 8; ++jj) {
            v4h h;
            h[0] = (_Float16)sreg[jj][0];
            h[1] = (_Float16)sreg[jj][1];
            h[2] = (_Float16)sreg[jj][2];
            h[3] = (_Float16)sreg[jj][3];
            *(v4h*)&bs[buf][(jg * 8 + jj) * KCH + s4] = h;
        }
    };

    stage_load(0);
    stage_store(0);
    __syncthreads();

    for (int it = 0; it < NCH; ++it) {
        const int buf = it & 1;
        if (it + 1 < NCH) stage_load(it + 1);        // long-latency loads issued first

        // ---- build 16x32 f16 A fragment: trig features generated in-register ----
        // A lane's 16 K-indices span <= 24 consecutive r -> at most 2 distinct i.
        const int rb = rseg + it * KCH + laneHalf;
        const unsigned i0 = (unsigned)rb / (unsigned)GRIDSZ;
        const unsigned i1 = (i0 + 1u < (unsigned)IDIM) ? (i0 + 1u) : (unsigned)(IDIM - 1);
        const float x0 = xrow[i0];
        const float x1 = xrow[i1];
        v16h a;
        #pragma unroll
        for (int h = 0; h < 16; ++h) {
            const unsigned r = (unsigned)(rb + (h & 7) + ((h & 8) << 1)); // K per ISA layout
            const unsigned i = r / (unsigned)GRIDSZ;     // magic-number div
            const unsigned g = r - i * (unsigned)GRIDSZ;
            const float xv = (i == i0) ? x0 : x1;        // v_cndmask, no vmem
            const float ph = xv * (float)(g + 1u) - shift;
            a[h] = (_Float16)__cosf(ph);                 // one v_cos_f32 per feature
        }

        // ---- sweep all 16 N-subtiles of the 256-wide C strip ----
        const _Float16* __restrict__ bbase = &bs[buf][0];
        #pragma unroll
        for (int n = 0; n < 16; ++n) {
            const _Float16* bp =
                bbase + (n * 16 + (lane & 15)) * KCH + (laneHalf << 1);
            v16h b = *(const v16h*)bp;                   // 2x ds_load_b128, fragment order
            acc[n] = __builtin_amdgcn_wmma_f32_16x16x32_f16(
                false, a, false, b, (short)0, acc[n], false, false);
        }

        if (it + 1 < NCH) stage_store(buf ^ 1);      // waits loadcnt here, after the sweep
        __syncthreads();
    }

    // ---- K-split reduction: atomically accumulate partial C into out ----
    const int mrow = mblk * MTILE + wave * 16 + laneHalf;  // + e below (C/D layout)
    const int ncol = lane & 15;
    #pragma unroll
    for (int n = 0; n < 16; ++n) {
        #pragma unroll
        for (int e = 0; e < 8; ++e) {
            atomicAdd(&out[(mrow + e) * ODIM + n * 16 + ncol], acc[n][e]);
        }
    }
}

extern "C" void kernel_launch(void* const* d_in, const int* in_sizes, int n_in,
                              void* d_out, int out_size, void* d_ws, size_t ws_size,
                              hipStream_t stream) {
    (void)in_sizes; (void)n_in; (void)d_ws; (void)ws_size; (void)out_size;
    const float* x    = (const float*)d_in[0];   // (1024, 256) f32
    const float* fc   = (const float*)d_in[1];   // (2, 256, 256, 300) f32
    const float* bias = (const float*)d_in[2];   // (1, 256) f32
    float* out = (float*)d_out;                  // (1024, 256) f32

    fkan_bias_init<<<dim3(BATCH), dim3(ODIM), 0, stream>>>(bias, out);
    fkan_wmma<<<dim3(MSPLIT * KSPLIT), dim3(NTHREADS), 0, stream>>>(x, fc, out);
}
